// MMD_lossclass3_5617817223859
// MI455X (gfx1250) — matrix operand order repaired and checked by reading, not compiled, not run on hardware
//
#include <hip/hip_runtime.h>
#include <hip/hip_bf16.h>

// ---------------------------------------------------------------------------
// MMD loss, fused for gfx1250 (MI455X).
//   N = 8192 total rows, D = 256, b = 4096.
//   Pass 1 (prep):   fp32 -> bf16 matrix in ws, row squared norms sq[i].
//   Pass 2 (stats):  column sums -> ||sum x||^2, sum(sq); closed-form
//                    sum(l2) = 2N*sumsq - 2*dot; bandwidth -> c = log2e/(16 bw).
//   Pass 3 (main):   bf16 WMMA GEMM tiles (upper triangle only, weight-2
//                    off-diagonal), fused epilogue: l2, single exp2 + square
//                    chain for the 5 RBF bandwidths, signed block means,
//                    fixed-order LDS reduction -> per-block partial.
//   Pass 4 (final):  deterministic reduction of 2048 partials -> d_out[0].
// ---------------------------------------------------------------------------

typedef __attribute__((ext_vector_type(16))) __bf16 v16bf;
typedef __attribute__((ext_vector_type(8)))  __bf16 v8bf;
typedef __attribute__((ext_vector_type(8)))  float  v8f;

#define NTOT 8192
#define DDIM 256
#define BHALF 4096

union AF { v16bf v; v8bf h[2]; };

// ---- Pass 1: convert to bf16, compute row squared norms -------------------
__global__ __launch_bounds__(256) void mmd_prep(const float* __restrict__ src,
                                                const float* __restrict__ tgt,
                                                __bf16* __restrict__ bmat,
                                                float* __restrict__ sqArr) {
  const int tid  = threadIdx.x;
  const int wave = tid >> 5, lane = tid & 31;
  const int row  = blockIdx.x * 8 + wave;
  const float* p = (row < BHALF) ? (src + (size_t)row * DDIM)
                                 : (tgt + (size_t)(row - BHALF) * DDIM);
  float q = 0.f;
#pragma unroll
  for (int e = 0; e < 8; ++e) {
    const int d = lane + e * 32;           // coalesced 128B per instruction
    const float x = p[d];
    bmat[(size_t)row * DDIM + d] = (__bf16)x;
    q += x * x;
  }
#pragma unroll
  for (int off = 16; off > 0; off >>= 1) q += __shfl_xor(q, off);
  if (lane == 0) sqArr[row] = q;
}

// ---- Pass 2: closed-form sum(l2) -> bandwidth scalar ----------------------
__global__ __launch_bounds__(256) void mmd_stats(const float* __restrict__ src,
                                                 const float* __restrict__ tgt,
                                                 const float* __restrict__ sqArr,
                                                 float* __restrict__ cw) {
  const int tid = threadIdx.x;
  __shared__ float red[256];
  // column sums of total (thread tid owns column tid; fully coalesced)
  float s = 0.f;
  for (int i = 0; i < BHALF; ++i) s += src[(size_t)i * DDIM + tid];
  for (int i = 0; i < BHALF; ++i) s += tgt[(size_t)i * DDIM + tid];
  red[tid] = s * s;
  __syncthreads();
#pragma unroll
  for (int st = 128; st > 0; st >>= 1) {
    if (tid < st) red[tid] += red[tid + st];
    __syncthreads();
  }
  const float dot = red[0];
  __syncthreads();
  // deterministic sum of sq[]
  float q = 0.f;
#pragma unroll
  for (int r = 0; r < NTOT / 256; ++r) q += sqArr[tid + r * 256];
  red[tid] = q;
  __syncthreads();
#pragma unroll
  for (int st = 128; st > 0; st >>= 1) {
    if (tid < st) red[tid] += red[tid + st];
    __syncthreads();
  }
  if (tid == 0) {
    const float sumsq = red[0];
    const float suml2 = 2.f * (float)NTOT * sumsq - 2.f * dot;
    const float bw = suml2 / ((float)NTOT * (float)NTOT - (float)NTOT) / 4.f;
    // terms are exp(-l2/(bw*2^i)), i=0..4 -> e^(2^(4-i)) with
    // e = exp2(-l2 * log2e / (16*bw))
    cw[0] = 1.4426950408889634f / (16.f * bw);
  }
}

// ---- Pass 3: WMMA GEMM + fused MMD epilogue (upper triangle) --------------
__global__ __launch_bounds__(256) void mmd_main(const __bf16* __restrict__ bm,
                                                const float* __restrict__ sqArr,
                                                const float* __restrict__ cw,
                                                float* __restrict__ partials) {
  const int tid = threadIdx.x;
  const int jb = blockIdx.x;            // 0..31  (256-wide column blocks)
  const int ib = blockIdx.y;            // 0..63  (128-tall row blocks)
  const int rowBlk = ib * 128, colBlk = jb * 256;
  const int blockLin = ib * gridDim.x + jb;
  if (colBlk + 256 <= rowBlk) {         // strictly below diagonal: no work
    if (tid == 0) partials[blockLin] = 0.f;
    return;
  }
  __shared__ float sqRow[128];
  __shared__ float sqCol[256];
  __shared__ float red[256];
  if (tid < 128) sqRow[tid] = sqArr[rowBlk + tid];
  sqCol[tid] = sqArr[colBlk + tid];
  __syncthreads();

  const int wave = tid >> 5, lane = tid & 31;
  const int wi = wave >> 2, wj = wave & 3;        // 2 x 4 waves
  const int rowBase = rowBlk + wi * 64;           // 64x64 per wave
  const int colBase = colBlk + wj * 64;
  const int m16 = lane & 15, khalf = lane >> 4;

  v8f acc[4][4];
#pragma unroll
  for (int ti = 0; ti < 4; ++ti)
#pragma unroll
    for (int tj = 0; tj < 4; ++tj) acc[ti][tj] = (v8f){};

  for (int kk = 0; kk < DDIM; kk += 32) {
    AF a[4], b[4];
    // A fragment (16x32 bf16): lanes 0-15 rows M, elems 0-7 = K[kk+8*khalf..),
    // elems 8-15 = K[kk+16+8*khalf..)  (ISA 7.12.2 16-bit A layout)
#pragma unroll
    for (int ti = 0; ti < 4; ++ti) {
      const int row = rowBase + ti * 16 + m16;
      const __bf16* ra = bm + (size_t)row * DDIM + kk + khalf * 8;
      a[ti].h[0] = *reinterpret_cast<const v8bf*>(ra);
      a[ti].h[1] = *reinterpret_cast<const v8bf*>(ra + 16);
    }
    // B fragment (32x16 bf16): B = A^T, so column n = row (colBase+n) of A;
    // lane n: elems 0-7 = K[kk+8*khalf..), elems 8-15 = K[kk+16+8*khalf..).
#pragma unroll
    for (int tj = 0; tj < 4; ++tj) {
      const int col = colBase + tj * 16 + m16;
      const __bf16* rb = bm + (size_t)col * DDIM + kk + khalf * 8;
      b[tj].h[0] = *reinterpret_cast<const v8bf*>(rb);
      b[tj].h[1] = *reinterpret_cast<const v8bf*>(rb + 16);
    }
#pragma unroll
    for (int ti = 0; ti < 4; ++ti)
#pragma unroll
      for (int tj = 0; tj < 4; ++tj)
        acc[ti][tj] = __builtin_amdgcn_wmma_f32_16x16x32_bf16(
            false, a[ti].v, false, b[tj].v, (short)0, acc[ti][tj], false,
            false);
  }

  // Fused epilogue: C layout -> (i,j); l2 -> 5-bandwidth RBF via one exp2.
  const float c = cw[0];
  float accs = 0.f;
#pragma unroll
  for (int ti = 0; ti < 4; ++ti) {
#pragma unroll
    for (int tj = 0; tj < 4; ++tj) {
      const int j = colBase + tj * 16 + m16;
      const float sqj = sqCol[wj * 64 + tj * 16 + m16];
#pragma unroll
      for (int v = 0; v < 8; ++v) {
        const int m = v + khalf * 8;          // C frag: VGPR v, lane half
        const int i = rowBase + ti * 16 + m;
        const float sqi = sqRow[wi * 64 + ti * 16 + m];
        const float g = acc[ti][tj][v];
        const float l2 = fmaxf(sqi + sqj - 2.f * g, 0.f);
        const float e1 = __builtin_amdgcn_exp2f(-l2 * c);  // exp(-l2/(16bw))
        const float e2 = e1 * e1;
        const float e4 = e2 * e2;
        const float e8 = e4 * e4;
        const float e16 = e8 * e8;
        const float ks = e1 + e2 + e4 + e8 + e16;
        const float w = (j > i) ? 2.f : ((j == i) ? 1.f : 0.f);
        const float sgn = ((i ^ j) & BHALF) ? -1.f : 1.f;
        accs += w * sgn * ks;
      }
    }
  }
  red[tid] = accs * 0x1p-24f;   // 1 / b^2, exact power of two
  __syncthreads();
#pragma unroll
  for (int st = 128; st > 0; st >>= 1) {
    if (tid < st) red[tid] += red[tid + st];
    __syncthreads();
  }
  if (tid == 0) partials[blockLin] = red[0];
}

// ---- Pass 4: deterministic final reduction --------------------------------
__global__ __launch_bounds__(256) void mmd_final(const float* __restrict__ partials,
                                                 float* __restrict__ out) {
  const int tid = threadIdx.x;
  __shared__ float red[256];
  float s = 0.f;
#pragma unroll
  for (int r = 0; r < 8; ++r) s += partials[tid + r * 256];   // 2048 partials
  red[tid] = s;
  __syncthreads();
#pragma unroll
  for (int st = 128; st > 0; st >>= 1) {
    if (tid < st) red[tid] += red[tid + st];
    __syncthreads();
  }
  if (tid == 0) out[0] = red[0];
}

extern "C" void kernel_launch(void* const* d_in, const int* in_sizes, int n_in,
                              void* d_out, int out_size, void* d_ws,
                              size_t ws_size, hipStream_t stream) {
  (void)in_sizes; (void)n_in; (void)out_size; (void)ws_size;
  const float* src = (const float*)d_in[0];
  const float* tgt = (const float*)d_in[1];
  float* wsf = (float*)d_ws;
  float* cw = wsf;                         // [0]  bandwidth scalar
  float* sqArr = wsf + 16;                 // 8192 row squared norms
  float* partials = wsf + 16 + NTOT;       // 2048 block partials
  __bf16* bmat = (__bf16*)(wsf + 16 + NTOT + 2048);  // 8192x256 bf16 (4 MB)
  float* out = (float*)d_out;

  mmd_prep<<<NTOT / 8, 256, 0, stream>>>(src, tgt, bmat, sqArr);
  mmd_stats<<<1, 256, 0, stream>>>(src, tgt, sqArr, cw);
  mmd_main<<<dim3(NTOT / 256, NTOT / 128), 256, 0, stream>>>(bmat, sqArr, cw,
                                                             partials);
  mmd_final<<<1, 256, 0, stream>>>(partials, out);
}
